// TransformerEncoder_231928234172
// MI455X (gfx1250) — matrix operand order repaired and checked by reading, not compiled, not run on hardware
//
#include <hip/hip_runtime.h>
#include <hip/hip_bf16.h>

// Problem constants (match reference)
constexpr int kB  = 4;
constexpr int kS  = 2048;
constexpr int kH  = 64;     // head dim == model-input dim
constexpr int kNH = 8;
constexpr int kL  = 2;
constexpr int kM  = 256;
constexpr int kD  = kH * kNH;  // 512

typedef __attribute__((ext_vector_type(16))) _Float16     v16h;
typedef __attribute__((ext_vector_type(8)))  float        v8f;
typedef __attribute__((ext_vector_type(4)))  unsigned int u32x4;
typedef int v4i __attribute__((vector_size(16)));  // matches async builtin param type

union Frag {
  v16h v;
  u32x4 u[2];
  unsigned short s[16];
};

#if __has_builtin(__builtin_amdgcn_global_load_async_to_lds_b128)
#define HAS_ASYNC_LDS 1
#endif

static __device__ __forceinline__ unsigned short f2h(float x) {
  union { _Float16 h; unsigned short u; } c;
  c.h = (_Float16)x;
  return c.u;
}

static __device__ __forceinline__ v8f wmma16(const Frag& a, const Frag& b, v8f c) {
  // D = A(16x32 f16) x B(32x16 f16) + C(16x16 f32)
  return __builtin_amdgcn_wmma_f32_16x16x32_f16(false, a.v, false, b.v, (short)0, c,
                                                false, false);
}

static __device__ __forceinline__ void wait_async_lds() {
#ifdef HAS_ASYNC_LDS
#if __has_builtin(__builtin_amdgcn_s_wait_asynccnt)
  __builtin_amdgcn_s_wait_asynccnt(0);
#else
  asm volatile("s_wait_asynccnt 0x0" ::: "memory");
#endif
#endif
}

// ---------------------------------------------------------------------------
// f32 -> f16 convert (optionally also copy f32)
// ---------------------------------------------------------------------------
__global__ __launch_bounds__(256) void cvt_f16_kernel(const float* __restrict__ in,
                                                      unsigned short* __restrict__ out16,
                                                      float* __restrict__ copy32, int n) {
  int i = blockIdx.x * blockDim.x + threadIdx.x;
  if (i < n) {
    float x = in[i];
    out16[i] = f2h(x);
    if (copy32) copy32[i] = x;
  }
}

// ---------------------------------------------------------------------------
// WMMA GEMM (compile-time K and NC; lda==K, ldb==ldc==NC for this network).
// Each wave computes a 16x64 output panel: one A fragment per K-step reused
// across 4 WMMAs; B fragments loaded into independent registers first so the
// backend can issue partial s_wait_loadcnt and overlap loads with WMMA.
// ---------------------------------------------------------------------------
template <int K, int NC>
__global__ __launch_bounds__(256) void wmma_gemm_kernel(
    const unsigned short* __restrict__ A,
    const unsigned short* __restrict__ Bm,
    const float* __restrict__ bias,
    unsigned short* __restrict__ outH, float* __restrict__ outF,
    int Mr, int relu) {
  constexpr int LDA = K;
  constexpr int LDB = NC;
  constexpr int LDC = NC;
  const int wid  = threadIdx.x >> 5;
  const int lane = threadIdx.x & 31;
  constexpr int tilesN = NC / 64;             // 64-wide N panels
  const int tiles = (Mr >> 4) * tilesN;
  const int tile = blockIdx.x * (blockDim.x >> 5) + wid;
  if (tile >= tiles) return;
  const int tm = tile / tilesN;
  const int tn = tile - tm * tilesN;
  const int m  = lane & 15;   // A row within tile / C column within tile
  const int hh = lane >> 4;   // half-wave select

  const v8f vzero = {0.f, 0.f, 0.f, 0.f, 0.f, 0.f, 0.f, 0.f};
  v8f c[4];
#pragma unroll
  for (int t = 0; t < 4; ++t) c[t] = vzero;

  const unsigned short* arow  = A + (size_t)(tm * 16 + m) * LDA;
  const unsigned short* bbase = Bm + tn * 64 + m;

#pragma unroll
  for (int k0 = 0; k0 < K; k0 += 32) {
    Frag a;
    a.u[0] = *(const u32x4*)(arow + k0 + hh * 8);
    a.u[1] = *(const u32x4*)(arow + k0 + 16 + hh * 8);
    const unsigned short* brow = bbase + (size_t)k0 * LDB;
    Frag bf[4];
#pragma unroll
    for (int t = 0; t < 4; ++t) {
#pragma unroll
      for (int s = 0; s < 8; ++s) {
        bf[t].s[s]     = brow[(size_t)(hh * 8 + s) * LDB + t * 16];
        bf[t].s[8 + s] = brow[(size_t)(16 + hh * 8 + s) * LDB + t * 16];
      }
    }
#pragma unroll
    for (int t = 0; t < 4; ++t) c[t] = wmma16(a, bf[t], c[t]);
  }

#pragma unroll
  for (int t = 0; t < 4; ++t) {
    const int col = tn * 64 + t * 16 + m;
    const float bv = bias ? bias[col] : 0.0f;
#pragma unroll
    for (int e = 0; e < 8; ++e) {
      float x = c[t][e] + bv;
      if (relu) x = fmaxf(x, 0.0f);
      const size_t off = (size_t)(tm * 16 + hh * 8 + e) * LDC + col;
      if (outF) outF[off] = x;
      if (outH) outH[off] = f2h(x);
    }
  }
}

// ---------------------------------------------------------------------------
// Flash attention: grid (B*NH, S/64), block 128 (4 waves, 16 query rows each)
// Q,K,V are (B*S, D) f16, head slice = [head*64, head*64+64).
// Masked entries get -1e12 (matching reference semantics incl. fully-masked rows).
// V chunk staged with GLOBAL_LOAD_ASYNC_TO_LDS_B128 when available.
// ---------------------------------------------------------------------------
__global__ __launch_bounds__(128) void flash_attn_kernel(
    const unsigned short* __restrict__ Q,
    const unsigned short* __restrict__ Kx,
    const unsigned short* __restrict__ V,
    const int* __restrict__ mask,
    unsigned short* __restrict__ O) {
  __shared__ unsigned short KT[64 * 32];    // transposed K chunk: KT[dim][key]
  __shared__ unsigned short Vs[32 * 64];    // V chunk row-major: Vs[key][dim]
  __shared__ unsigned short Pw[4][16 * 32]; // per-wave P tile, row-major [row][key]
  __shared__ int kval[32];

  const int b    = blockIdx.x / kNH;
  const int head = blockIdx.x - b * kNH;
  const int wid  = threadIdx.x >> 5;
  const int lane = threadIdx.x & 31;
  const int tid  = threadIdx.x;
  const int q0   = blockIdx.y * 64 + wid * 16;
  const int m    = lane & 15;
  const int hh   = lane >> 4;

  const v8f vzero = {0.f, 0.f, 0.f, 0.f, 0.f, 0.f, 0.f, 0.f};

  // query-valid flags for the 8 rows this lane's half-wave owns
  int qv[8];
#pragma unroll
  for (int e = 0; e < 8; ++e)
    qv[e] = (mask[b * kS + q0 + hh * 8 + e] != 0) ? 1 : 0;

  // hoist Q fragments (reduction dim = head dim 64 -> 2 chunks of 32)
  Frag qf[2];
  {
    const unsigned short* qrow = Q + (size_t)(b * kS + q0 + m) * kD + head * 64;
#pragma unroll
    for (int kk = 0; kk < 2; ++kk) {
      qf[kk].u[0] = *(const u32x4*)(qrow + kk * 32 + hh * 8);
      qf[kk].u[1] = *(const u32x4*)(qrow + kk * 32 + 16 + hh * 8);
    }
  }

  v8f oc[4];
#pragma unroll
  for (int t = 0; t < 4; ++t) oc[t] = vzero;
  float rowmax[8], rowsum[8];
#pragma unroll
  for (int e = 0; e < 8; ++e) { rowmax[e] = -1e30f; rowsum[e] = 0.0f; }

  const float scale = 0.125f;  // 1/sqrt(64)

  for (int kb = 0; kb < kS; kb += 32) {
    __syncthreads();  // protect LDS reuse from previous iteration
    // Cooperative staging: 32 keys x 64 dims of K (transposed) and V (row-major)
#pragma unroll
    for (int i0 = 0; i0 < 2; ++i0) {
      const int i = tid + i0 * 128;
      const int key = i >> 3, c8 = i & 7;
      const size_t gidx = (size_t)(b * kS + kb + key) * kD + head * 64 + c8 * 8;
      union { u32x4 u; unsigned short s[8]; } kd;
      kd.u = *(const u32x4*)(Kx + gidx);
#pragma unroll
      for (int j = 0; j < 8; ++j) KT[(c8 * 8 + j) * 32 + key] = kd.s[j];
#ifdef HAS_ASYNC_LDS
      __builtin_amdgcn_global_load_async_to_lds_b128(
          (__attribute__((address_space(1))) v4i*)(V + gidx),
          (__attribute__((address_space(3))) v4i*)(&Vs[key * 64 + c8 * 8]), 0, 0);
#else
      *(u32x4*)(&Vs[key * 64 + c8 * 8]) = *(const u32x4*)(V + gidx);
#endif
      // prefetch next chunk while this one is consumed
      if (kb + 32 < kS) {
        __builtin_prefetch(Kx + gidx + (size_t)32 * kD, 0, 1);
        __builtin_prefetch(V + gidx + (size_t)32 * kD, 0, 1);
      }
    }
    if (tid < 32) kval[tid] = (mask[b * kS + kb + tid] != 0) ? 1 : 0;
    wait_async_lds();
    __syncthreads();

    // Scores: S = Q(16x64) @ K^T(64x32) as two 16x16 n-tiles
    v8f c0 = vzero, c1 = vzero;
#pragma unroll
    for (int kk = 0; kk < 2; ++kk) {
      Frag b0, b1;
#pragma unroll
      for (int s = 0; s < 8; ++s) {
        b0.s[s]     = KT[(kk * 32 + hh * 8 + s) * 32 + m];
        b0.s[8 + s] = KT[(kk * 32 + 16 + hh * 8 + s) * 32 + m];
        b1.s[s]     = KT[(kk * 32 + hh * 8 + s) * 32 + 16 + m];
        b1.s[8 + s] = KT[(kk * 32 + 16 + hh * 8 + s) * 32 + 16 + m];
      }
      c0 = wmma16(qf[kk], b0, c0);
      c1 = wmma16(qf[kk], b1, c1);
    }

    const int kv0 = kval[m];
    const int kv1 = kval[16 + m];
    unsigned short* pw = &Pw[wid][0];
    float alpha8[8];
#pragma unroll
    for (int e = 0; e < 8; ++e) {
      float s0 = (qv[e] && kv0) ? c0[e] * scale : -1e12f;
      float s1 = (qv[e] && kv1) ? c1[e] * scale : -1e12f;
      float mx = fmaxf(s0, s1);
#pragma unroll
      for (int o = 8; o; o >>= 1) mx = fmaxf(mx, __shfl_xor(mx, o, 16));
      const float mnew  = fmaxf(rowmax[e], mx);
      const float alpha = __expf(rowmax[e] - mnew);
      const float p0 = __expf(s0 - mnew);
      const float p1 = __expf(s1 - mnew);
      float ps = p0 + p1;
#pragma unroll
      for (int o = 8; o; o >>= 1) ps += __shfl_xor(ps, o, 16);
      rowsum[e] = rowsum[e] * alpha + ps;
      rowmax[e] = mnew;
      alpha8[e] = alpha;
      pw[(hh * 8 + e) * 32 + m]      = f2h(p0);
      pw[(hh * 8 + e) * 32 + 16 + m] = f2h(p1);
    }
#pragma unroll
    for (int t = 0; t < 4; ++t)
#pragma unroll
      for (int e = 0; e < 8; ++e) oc[t][e] *= alpha8[e];

    // O += P(16x32) @ V(32x64): A-layout read of P from LDS, V rows from LDS
    Frag pf;
    const unsigned short* pr = pw + m * 32;
    pf.u[0] = *(const u32x4*)(pr + hh * 8);
    pf.u[1] = *(const u32x4*)(pr + 16 + hh * 8);
    Frag vf[4];
#pragma unroll
    for (int t = 0; t < 4; ++t) {
#pragma unroll
      for (int s = 0; s < 8; ++s) {
        vf[t].s[s]     = Vs[(hh * 8 + s) * 64 + t * 16 + m];
        vf[t].s[8 + s] = Vs[(16 + hh * 8 + s) * 64 + t * 16 + m];
      }
    }
#pragma unroll
    for (int t = 0; t < 4; ++t) oc[t] = wmma16(pf, vf[t], oc[t]);
  }

#pragma unroll
  for (int e = 0; e < 8; ++e) {
    const float rinv = 1.0f / rowsum[e];
#pragma unroll
    for (int t = 0; t < 4; ++t) {
      const size_t off = (size_t)(b * kS + q0 + hh * 8 + e) * kD + head * 64 + t * 16 + m;
      O[off] = f2h(oc[t][e] * rinv);
    }
  }
}

// ---------------------------------------------------------------------------
// Fused residual-add + LayerNorm over 64 features; one wave per row.
// ---------------------------------------------------------------------------
__global__ __launch_bounds__(256) void add_ln_kernel(
    const float* __restrict__ a, const float* __restrict__ res,
    const float* __restrict__ g, const float* __restrict__ be,
    float* __restrict__ out32, unsigned short* __restrict__ out16, int rows) {
  const int wid  = threadIdx.x >> 5;
  const int lane = threadIdx.x & 31;
  const int row  = blockIdx.x * (blockDim.x >> 5) + wid;
  if (row >= rows) return;
  const size_t base = (size_t)row * 64;
  float x0 = a[base + lane]      + res[base + lane];
  float x1 = a[base + 32 + lane] + res[base + 32 + lane];
  float s = x0 + x1;
#pragma unroll
  for (int o = 16; o; o >>= 1) s += __shfl_xor(s, o, 32);
  const float mu = s * (1.0f / 64.0f);
  const float d0 = x0 - mu, d1 = x1 - mu;
  float v = d0 * d0 + d1 * d1;
#pragma unroll
  for (int o = 16; o; o >>= 1) v += __shfl_xor(v, o, 32);
  const float rstd = rsqrtf(v * (1.0f / 64.0f) + 1e-5f);
  const float y0 = d0 * rstd * g[lane]      + be[lane];
  const float y1 = d1 * rstd * g[lane + 32] + be[lane + 32];
  out32[base + lane]      = y0;
  out32[base + 32 + lane] = y1;
  if (out16) {
    out16[base + lane]      = f2h(y0);
    out16[base + 32 + lane] = f2h(y1);
  }
}

// ---------------------------------------------------------------------------
extern "C" void kernel_launch(void* const* d_in, const int* in_sizes, int n_in,
                              void* d_out, int out_size, void* d_ws, size_t ws_size,
                              hipStream_t stream) {
  (void)in_sizes; (void)n_in; (void)out_size; (void)ws_size;
  const float* x    = (const float*)d_in[0];
  const int*   mask = (const int*)d_in[1];
  const float* Wq = (const float*)d_in[2];  const float* bq  = (const float*)d_in[3];
  const float* Wk = (const float*)d_in[4];  const float* bk  = (const float*)d_in[5];
  const float* Wv = (const float*)d_in[6];  const float* bv  = (const float*)d_in[7];
  const float* Wo = (const float*)d_in[8];  const float* bo  = (const float*)d_in[9];
  const float* W1 = (const float*)d_in[10]; const float* b1  = (const float*)d_in[11];
  const float* W2 = (const float*)d_in[12]; const float* b2  = (const float*)d_in[13];
  const float* g1 = (const float*)d_in[14]; const float* be1 = (const float*)d_in[15];
  const float* g2 = (const float*)d_in[16]; const float* be2 = (const float*)d_in[17];

  const int rows = kB * kS;  // 8192

  char* p = (char*)d_ws;
  auto alloc = [&](size_t bytes) -> char* {
    char* r = p;
    p += (bytes + 255) & ~(size_t)255;
    return r;
  };
  float*          cur32 = (float*)alloc((size_t)rows * kH * 4);
  unsigned short* cur16 = (unsigned short*)alloc((size_t)rows * kH * 2);
  unsigned short* Q16   = (unsigned short*)alloc((size_t)rows * kD * 2);
  unsigned short* K16   = (unsigned short*)alloc((size_t)rows * kD * 2);
  unsigned short* V16   = (unsigned short*)alloc((size_t)rows * kD * 2);
  unsigned short* O16   = (unsigned short*)alloc((size_t)rows * kD * 2);
  float*          tmp32 = (float*)alloc((size_t)rows * kH * 4);
  unsigned short* A116  = (unsigned short*)alloc((size_t)rows * kM * 2);
  unsigned short* Wq16  = (unsigned short*)alloc((size_t)kL * kH * kD * 2);
  unsigned short* Wk16  = (unsigned short*)alloc((size_t)kL * kH * kD * 2);
  unsigned short* Wv16  = (unsigned short*)alloc((size_t)kL * kH * kD * 2);
  unsigned short* Wo16  = (unsigned short*)alloc((size_t)kL * kD * kH * 2);
  unsigned short* W116  = (unsigned short*)alloc((size_t)kL * kH * kM * 2);
  unsigned short* W216  = (unsigned short*)alloc((size_t)kL * kM * kH * 2);

  auto cvt = [&](const float* in, unsigned short* o16, float* c32, int n) {
    cvt_f16_kernel<<<(n + 255) / 256, 256, 0, stream>>>(in, o16, c32, n);
  };
  cvt(x,  cur16, cur32, rows * kH);
  cvt(Wq, Wq16, nullptr, kL * kH * kD);
  cvt(Wk, Wk16, nullptr, kL * kH * kD);
  cvt(Wv, Wv16, nullptr, kL * kH * kD);
  cvt(Wo, Wo16, nullptr, kL * kD * kH);
  cvt(W1, W116, nullptr, kL * kH * kM);
  cvt(W2, W216, nullptr, kL * kM * kH);

  const int blkQKV = ((rows / 16) * (kD / 64) + 7) / 8;   // N=512
  const int blkOP  = ((rows / 16) * (kH / 64) + 7) / 8;   // N=64
  const int blkF1  = ((rows / 16) * (kM / 64) + 7) / 8;   // N=256
  const int blkF2  = blkOP;                               // N=64

  for (int l = 0; l < kL; ++l) {
    // QKV projections: (8192 x 64) @ (64 x 512)
    wmma_gemm_kernel<kH, kD><<<blkQKV, 256, 0, stream>>>(
        cur16, Wq16 + (size_t)l * kH * kD, bq + (size_t)l * kD, Q16, nullptr, rows, 0);
    wmma_gemm_kernel<kH, kD><<<blkQKV, 256, 0, stream>>>(
        cur16, Wk16 + (size_t)l * kH * kD, bk + (size_t)l * kD, K16, nullptr, rows, 0);
    wmma_gemm_kernel<kH, kD><<<blkQKV, 256, 0, stream>>>(
        cur16, Wv16 + (size_t)l * kH * kD, bv + (size_t)l * kD, V16, nullptr, rows, 0);
    // Flash attention -> O16 (B*S x D, head-major columns)
    flash_attn_kernel<<<dim3(kB * kNH, kS / 64), 128, 0, stream>>>(Q16, K16, V16, mask, O16);
    // Output projection: (8192 x 512) @ (512 x 64)
    wmma_gemm_kernel<kD, kH><<<blkOP, 256, 0, stream>>>(
        O16, Wo16 + (size_t)l * kD * kH, bo + (size_t)l * kH, nullptr, tmp32, rows, 0);
    // Residual + LN1
    add_ln_kernel<<<rows / 8, 256, 0, stream>>>(tmp32, cur32, g1 + (size_t)l * kH,
                                                be1 + (size_t)l * kH, cur32, cur16, rows);
    // FFN1 with ReLU: (8192 x 64) @ (64 x 256)
    wmma_gemm_kernel<kH, kM><<<blkF1, 256, 0, stream>>>(
        cur16, W116 + (size_t)l * kH * kM, b1 + (size_t)l * kM, A116, nullptr, rows, 1);
    // FFN2: (8192 x 256) @ (256 x 64)
    wmma_gemm_kernel<kM, kH><<<blkF2, 256, 0, stream>>>(
        A116, W216 + (size_t)l * kM * kH, b2 + (size_t)l * kH, nullptr, tmp32, rows, 0);
    // Residual + LN2 (last layer writes final output)
    float* lnout = (l == kL - 1) ? (float*)d_out : cur32;
    add_ln_kernel<<<rows / 8, 256, 0, stream>>>(tmp32, cur32, g2 + (size_t)l * kH,
                                                be2 + (size_t)l * kH, lnout, cur16, rows);
  }
}
